// Net_52183852646955
// MI455X (gfx1250) — matrix operand order repaired and checked by reading, not compiled, not run on hardware
//
#include <hip/hip_runtime.h>
#include <hip/hip_bf16.h>
#include <stdint.h>

#define STU_N  50000
#define EXER_N 20000
#define KDIM   1024
#define BATCH  256
#define SAMP   32
#define L1DIM  512
#define L2DIM  256

#define SEED_STU 101u
#define SEED_KD  202u
#define SEED_ED  303u
#define SEED_W1  11u
#define SEED_B1  12u
#define SEED_W2  21u
#define SEED_B2  22u
#define SEED_W3  31u
#define SEED_B3  32u

#define LOG2E  1.44269504089f
#define LN2    0.69314718056f

typedef _Float16 v16h  __attribute__((ext_vector_type(16)));
typedef _Float16 half8 __attribute__((ext_vector_type(8)));
typedef float    v8f   __attribute__((ext_vector_type(8)));

__device__ __forceinline__ uint32_t h32(uint32_t x) {
    x ^= x >> 16; x *= 0x7feb352du;
    x ^= x >> 15; x *= 0x846ca68bu;
    x ^= x >> 16; return x;
}

// native-unit helpers: single v_exp/v_log/v_rcp/v_sqrt/v_cos, branchless
__device__ __forceinline__ float fexp(float x) {           // e^x
    return __builtin_amdgcn_exp2f(x * LOG2E);
}
__device__ __forceinline__ float splus(float x) {          // softplus, stable
    float t = __builtin_amdgcn_exp2f(-fabsf(x) * LOG2E);
    return fmaxf(x, 0.0f) + LN2 * __builtin_amdgcn_logf(1.0f + t);
}
__device__ __forceinline__ float sigm(float x) {
    return __builtin_amdgcn_rcpf(1.0f + __builtin_amdgcn_exp2f(-x * LOG2E));
}

// deterministic hash-based standard normal: ONE 32-bit hash -> two 16-bit
// uniforms -> Box-Muller (cos branch). v_cos_f32 computes cos(2*pi*x)
// natively, so the [0,1) uniform needs no radian scaling.
__device__ __forceinline__ float nrand(uint32_t seed, uint32_t idx) {
    uint32_t a = h32(idx + seed * 0x9E3779B9u);
    float u1 = (float)((a & 0xFFFFu) + 1u) * (1.0f / 65536.0f);  // (0,1]
    float u2 = (float)(a >> 16) * (1.0f / 65536.0f);             // [0,1)
    float r  = __builtin_amdgcn_sqrtf(fmaxf(-2.0f * LN2 * __builtin_amdgcn_logf(u1), 0.0f));
    return r * __builtin_amdgcn_cosf(u2);                        // cos(2*pi*u2)
}

// ---------------------------------------------------------------------------
// zero the scalar KL accumulator slots
// ---------------------------------------------------------------------------
__global__ void zero_tail_kernel(float* out) {
    if (threadIdx.x < 3) out[8192 + threadIdx.x] = 0.0f;
}

// ---------------------------------------------------------------------------
// Layer 1: build x[s,k] in LDS (f16), then batched sampled-weight GEMM
//   per block b:  [32 x 1024] @ [1024 x 512]  -> sigmoid -> x2 (f16)
// ---------------------------------------------------------------------------
__global__ __launch_bounds__(256) void layer1_kernel(
    const int* __restrict__ stu_id, const int* __restrict__ exer_id,
    const float* __restrict__ kp,
    const float* __restrict__ student_mean, const float* __restrict__ student_std_eta,
    const float* __restrict__ k_diff_mean,  const float* __restrict__ k_diff_std_eta,
    const float* __restrict__ e_disc_mean,  const float* __restrict__ e_disc_std_eta,
    const float* __restrict__ lam1_stu, const float* __restrict__ lam2_stu,
    const float* __restrict__ lam1_exer, const float* __restrict__ lam2_exer,
    const float* __restrict__ stu_cnt, const float* __restrict__ exer_cnt,
    const float* __restrict__ w_mu1, const float* __restrict__ w_eta1,
    const float* __restrict__ b_mu1, const float* __restrict__ b_eta1,
    _Float16* __restrict__ x2)
{
    __shared__ __attribute__((aligned(16))) _Float16 xs[SAMP][KDIM + 8];

    const int b   = blockIdx.x;
    const int tid = threadIdx.x;
    const int sid = stu_id[b];
    const int eid = exer_id[b];

    const float l1s = splus(lam1_stu[0]),  l2s = splus(lam2_stu[0]);
    const float l1e = splus(lam1_exer[0]), l2e = splus(lam2_exer[0]);
    const float exstd = l1e * fexp(-l2e * exer_cnt[eid]);
    const float edm = e_disc_mean[eid];
    const float eds = exstd + splus(e_disc_std_eta[eid]);

    // ---- Phase A: activations into LDS (e_disc hoisted per sample s) -----
    for (int s = 0; s < SAMP; ++s) {
        uint32_t sb = (uint32_t)s * BATCH + (uint32_t)b;          // eps shape (S,B,*)
        float ed = sigm(edm + eds * nrand(SEED_ED, sb));
        #pragma unroll
        for (int j = 0; j < KDIM / 256; ++j) {
            int k = j * 256 + tid;
            float sm = student_mean[sid * KDIM + k];
            float ss = l1s * fexp(-l2s * stu_cnt[sid * KDIM + k])
                     + splus(student_std_eta[sid * KDIM + k]);
            float km = k_diff_mean[eid * KDIM + k];
            float ks = exstd + splus(k_diff_std_eta[eid * KDIM + k]);
            float e1 = nrand(SEED_STU, sb * KDIM + k);
            float e2 = nrand(SEED_KD,  sb * KDIM + k);
            float x  = ed * (sigm(sm + ss * e1) - sigm(km + ks * e2)) * kp[b * KDIM + k];
            xs[s][k] = (_Float16)x;
        }
    }
    __syncthreads();

    // ---- Phase B: WMMA GEMM ----------------------------------------------
    const int wave  = tid >> 5;
    const int lane  = tid & 31;
    const int halfL = lane >> 4;     // 0: lanes 0-15, 1: lanes 16-31
    const int n16   = lane & 15;
    const int oW    = wave * 64;     // this wave's 64 output columns

    v8f acc[2][4];
    #pragma unroll
    for (int mt = 0; mt < 2; ++mt)
        #pragma unroll
        for (int nt = 0; nt < 4; ++nt) acc[mt][nt] = {};

    for (int kb = 0; kb < KDIM; kb += 32) {
        // A fragments: 16x32 f16 tiles, lane m=lane%16, two contiguous 8-half runs
        v16h afrag[2];
        #pragma unroll
        for (int mt = 0; mt < 2; ++mt) {
            int m = mt * 16 + n16;
            half8 lo = *(const half8*)&xs[m][kb + halfL * 16];
            half8 hi = *(const half8*)&xs[m][kb + halfL * 16 + 8];
            v16h a;
            #pragma unroll
            for (int e = 0; e < 8; ++e) { a[e] = lo[e]; a[8 + e] = hi[e]; }
            afrag[mt] = a;
        }
        // B fragments: sample w[b,o,i] on the fly (16 contiguous i per lane)
        #pragma unroll
        for (int nt = 0; nt < 4; ++nt) {
            int o  = oW + nt * 16 + n16;
            int i0 = kb + halfL * 16;
            const float4* pm = (const float4*)&w_mu1 [o * KDIM + i0];
            const float4* pe = (const float4*)&w_eta1[o * KDIM + i0];
            uint32_t base = ((uint32_t)b * L1DIM + (uint32_t)o) * KDIM + (uint32_t)i0;
            v16h bf;
            #pragma unroll
            for (int q = 0; q < 4; ++q) {
                float4 m4 = pm[q];
                float4 e4 = pe[q];
                bf[q*4+0] = (_Float16)fexp(m4.x + (1e-6f + splus(e4.x)) * nrand(SEED_W1, base + q*4 + 0));
                bf[q*4+1] = (_Float16)fexp(m4.y + (1e-6f + splus(e4.y)) * nrand(SEED_W1, base + q*4 + 1));
                bf[q*4+2] = (_Float16)fexp(m4.z + (1e-6f + splus(e4.z)) * nrand(SEED_W1, base + q*4 + 2));
                bf[q*4+3] = (_Float16)fexp(m4.w + (1e-6f + splus(e4.w)) * nrand(SEED_W1, base + q*4 + 3));
            }
            acc[0][nt] = __builtin_amdgcn_wmma_f32_16x16x32_f16(false, afrag[0], false, bf, (short)0, acc[0][nt], false, false);
            acc[1][nt] = __builtin_amdgcn_wmma_f32_16x16x32_f16(false, afrag[1], false, bf, (short)0, acc[1][nt], false, false);
        }
    }

    // ---- Epilogue: sampled bias + sigmoid, store f16 ---------------------
    #pragma unroll
    for (int nt = 0; nt < 4; ++nt) {
        int o = oW + nt * 16 + n16;
        float bias = b_mu1[o] + (1e-6f + splus(b_eta1[o])) * nrand(SEED_B1, (uint32_t)b * L1DIM + o);
        #pragma unroll
        for (int mt = 0; mt < 2; ++mt) {
            #pragma unroll
            for (int r = 0; r < 8; ++r) {
                int s = mt * 16 + halfL * 8 + r;   // C layout: M = r + 8*halfL
                float v = sigm(acc[mt][nt][r] + bias);
                x2[((uint32_t)b * SAMP + s) * L1DIM + o] = (_Float16)v;
            }
        }
    }
}

// ---------------------------------------------------------------------------
// Layer 2: per block b:  [32 x 512] @ [512 x 256] -> sigmoid -> x3 (f16)
// Activation tile staged global->LDS with CDNA5 async load-to-LDS (ASYNCcnt).
// ---------------------------------------------------------------------------
__global__ __launch_bounds__(256) void layer2_kernel(
    const _Float16* __restrict__ x2,
    const float* __restrict__ w_mu2, const float* __restrict__ w_eta2,
    const float* __restrict__ b_mu2, const float* __restrict__ b_eta2,
    _Float16* __restrict__ x3)
{
    __shared__ __attribute__((aligned(16))) _Float16 xs[SAMP][L1DIM + 8];

    const int b   = blockIdx.x;
    const int tid = threadIdx.x;

    // ---- async copy of the [32 x 512] f16 tile into LDS ------------------
    {
        const unsigned long long src =
            (unsigned long long)(uintptr_t)(x2 + (uint32_t)b * SAMP * L1DIM);
        #pragma unroll
        for (int r = 0; r < (SAMP * L1DIM) / (256 * 8); ++r) {  // 8 iters of b128
            uint32_t v = (uint32_t)(r * 256 + tid);             // 16-byte chunk id
            uint32_t s = v >> 6;                                // 64 chunks per row
            uint32_t c = v & 63u;
            uint32_t lds_off = (s * (L1DIM + 8) + c * 8) * (uint32_t)sizeof(_Float16);
            uint32_t goff    = v * 16u;                         // tile rows are dense
            asm volatile("global_load_async_to_lds_b128 %0, %1, %2"
                         :: "v"(lds_off), "v"(goff), "s"(src)
                         : "memory");
        }
        asm volatile("s_wait_asynccnt 0x0" ::: "memory");
    }
    __syncthreads();

    const int wave  = tid >> 5;
    const int lane  = tid & 31;
    const int halfL = lane >> 4;
    const int n16   = lane & 15;
    const int oW    = wave * 32;            // 2 N-tiles per wave

    v8f acc[2][2];
    #pragma unroll
    for (int mt = 0; mt < 2; ++mt)
        #pragma unroll
        for (int nt = 0; nt < 2; ++nt) acc[mt][nt] = {};

    for (int kb = 0; kb < L1DIM; kb += 32) {
        v16h afrag[2];
        #pragma unroll
        for (int mt = 0; mt < 2; ++mt) {
            int m = mt * 16 + n16;
            half8 lo = *(const half8*)&xs[m][kb + halfL * 16];
            half8 hi = *(const half8*)&xs[m][kb + halfL * 16 + 8];
            v16h a;
            #pragma unroll
            for (int e = 0; e < 8; ++e) { a[e] = lo[e]; a[8 + e] = hi[e]; }
            afrag[mt] = a;
        }
        #pragma unroll
        for (int nt = 0; nt < 2; ++nt) {
            int o  = oW + nt * 16 + n16;
            int i0 = kb + halfL * 16;
            const float4* pm = (const float4*)&w_mu2 [o * L1DIM + i0];
            const float4* pe = (const float4*)&w_eta2[o * L1DIM + i0];
            uint32_t base = ((uint32_t)b * L2DIM + (uint32_t)o) * L1DIM + (uint32_t)i0;
            v16h bf;
            #pragma unroll
            for (int q = 0; q < 4; ++q) {
                float4 m4 = pm[q];
                float4 e4 = pe[q];
                bf[q*4+0] = (_Float16)fexp(m4.x + (1e-6f + splus(e4.x)) * nrand(SEED_W2, base + q*4 + 0));
                bf[q*4+1] = (_Float16)fexp(m4.y + (1e-6f + splus(e4.y)) * nrand(SEED_W2, base + q*4 + 1));
                bf[q*4+2] = (_Float16)fexp(m4.z + (1e-6f + splus(e4.z)) * nrand(SEED_W2, base + q*4 + 2));
                bf[q*4+3] = (_Float16)fexp(m4.w + (1e-6f + splus(e4.w)) * nrand(SEED_W2, base + q*4 + 3));
            }
            acc[0][nt] = __builtin_amdgcn_wmma_f32_16x16x32_f16(false, afrag[0], false, bf, (short)0, acc[0][nt], false, false);
            acc[1][nt] = __builtin_amdgcn_wmma_f32_16x16x32_f16(false, afrag[1], false, bf, (short)0, acc[1][nt], false, false);
        }
    }

    #pragma unroll
    for (int nt = 0; nt < 2; ++nt) {
        int o = oW + nt * 16 + n16;
        float bias = b_mu2[o] + (1e-6f + splus(b_eta2[o])) * nrand(SEED_B2, (uint32_t)b * L2DIM + o);
        #pragma unroll
        for (int mt = 0; mt < 2; ++mt) {
            #pragma unroll
            for (int r = 0; r < 8; ++r) {
                int s = mt * 16 + halfL * 8 + r;
                float v = sigm(acc[mt][nt][r] + bias);
                x3[((uint32_t)b * SAMP + s) * L2DIM + o] = (_Float16)v;
            }
        }
    }
}

// ---------------------------------------------------------------------------
// Layer 3: per block b:  [32 x 256] @ [256 x 1] -> sigmoid -> out[s, b]
// ---------------------------------------------------------------------------
__global__ __launch_bounds__(256) void layer3_kernel(
    const _Float16* __restrict__ x3,
    const float* __restrict__ w_mu3, const float* __restrict__ w_eta3,
    const float* __restrict__ b_mu3, const float* __restrict__ b_eta3,
    float* __restrict__ out)
{
    __shared__ float w3[L2DIM];
    const int b   = blockIdx.x;
    const int tid = threadIdx.x;

    w3[tid] = fexp(w_mu3[tid] + (1e-6f + splus(w_eta3[tid]))
                   * nrand(SEED_W3, (uint32_t)b * L2DIM + tid));
    __syncthreads();

    if (tid < SAMP) {
        float bias = b_mu3[0] + (1e-6f + splus(b_eta3[0])) * nrand(SEED_B3, (uint32_t)b);
        float acc = bias;
        const _Float16* row = &x3[((uint32_t)b * SAMP + tid) * L2DIM];
        #pragma unroll 8
        for (int i = 0; i < L2DIM; ++i) acc += (float)row[i] * w3[i];
        out[tid * BATCH + b] = sigm(acc);
    }
}

// ---------------------------------------------------------------------------
// KL(N(mu, 1e-6+softplus(eta)) || N(0,1)) summed -> atomicAdd(slot)
// ---------------------------------------------------------------------------
__global__ void kl_gauss_kernel(const float* __restrict__ mu,
                                const float* __restrict__ eta,
                                int n, float* slot)
{
    __shared__ float red[256];
    float s = 0.0f;
    for (int i = blockIdx.x * blockDim.x + threadIdx.x; i < n;
         i += gridDim.x * blockDim.x) {
        float st = 1e-6f + splus(eta[i]);
        float m  = mu[i];
        s += 0.5f * (st * st + m * m - 1.0f) - LN2 * __builtin_amdgcn_logf(st);
    }
    red[threadIdx.x] = s;
    __syncthreads();
    for (int off = 128; off > 0; off >>= 1) {
        if (threadIdx.x < off) red[threadIdx.x] += red[threadIdx.x + off];
        __syncthreads();
    }
    if (threadIdx.x == 0) atomicAdd(slot, red[0]);
}

// KL over full tables with count-dependent std (bandwidth-bound stream)
__global__ void kl_diag_kernel(const float* __restrict__ mu,
                               const float* __restrict__ eta,
                               const float* __restrict__ cnt,
                               const float* __restrict__ lam1,
                               const float* __restrict__ lam2,
                               int n, int cnt_div, float* slot)
{
    __shared__ float red[256];
    const float l1 = splus(lam1[0]), l2 = splus(lam2[0]);
    float s = 0.0f;
    for (int i = blockIdx.x * blockDim.x + threadIdx.x; i < n;
         i += gridDim.x * blockDim.x) {
        float st = l1 * fexp(-l2 * cnt[i / cnt_div]) + splus(eta[i]) + 1e-5f;
        float m  = mu[i];
        s += 0.5f * (st * st + m * m - 1.0f) - LN2 * __builtin_amdgcn_logf(st);
    }
    red[threadIdx.x] = s;
    __syncthreads();
    for (int off = 128; off > 0; off >>= 1) {
        if (threadIdx.x < off) red[threadIdx.x] += red[threadIdx.x + off];
        __syncthreads();
    }
    if (threadIdx.x == 0) atomicAdd(slot, red[0]);
}

// ---------------------------------------------------------------------------
extern "C" void kernel_launch(void* const* d_in, const int* in_sizes, int n_in,
                              void* d_out, int out_size, void* d_ws, size_t ws_size,
                              hipStream_t stream) {
    const int*   stu_id          = (const int*)  d_in[0];
    const int*   exer_id         = (const int*)  d_in[1];
    const float* kp              = (const float*)d_in[2];
    // d_in[3] = y (unused)
    const float* student_mean    = (const float*)d_in[4];
    const float* student_std_eta = (const float*)d_in[5];
    const float* k_diff_mean     = (const float*)d_in[6];
    const float* k_diff_std_eta  = (const float*)d_in[7];
    const float* e_disc_mean     = (const float*)d_in[8];
    const float* e_disc_std_eta  = (const float*)d_in[9];
    const float* lam1_stu        = (const float*)d_in[10];
    const float* lam2_stu        = (const float*)d_in[11];
    const float* lam1_exer       = (const float*)d_in[12];
    const float* lam2_exer       = (const float*)d_in[13];
    const float* stu_cnt         = (const float*)d_in[14];
    const float* exer_cnt        = (const float*)d_in[15];
    const float* w_mu1           = (const float*)d_in[16];
    const float* w_eta1          = (const float*)d_in[17];
    const float* b_mu1           = (const float*)d_in[18];
    const float* b_eta1          = (const float*)d_in[19];
    const float* w_mu2           = (const float*)d_in[20];
    const float* w_eta2          = (const float*)d_in[21];
    const float* b_mu2           = (const float*)d_in[22];
    const float* b_eta2          = (const float*)d_in[23];
    const float* w_mu3           = (const float*)d_in[24];
    const float* w_eta3          = (const float*)d_in[25];
    const float* b_mu3           = (const float*)d_in[26];
    const float* b_eta3          = (const float*)d_in[27];

    float* out = (float*)d_out;

    _Float16* x2 = (_Float16*)d_ws;                                             // 8 MB
    _Float16* x3 = (_Float16*)((char*)d_ws + (size_t)BATCH * SAMP * L1DIM * 2); // +4 MB

    zero_tail_kernel<<<1, 32, 0, stream>>>(out);

    layer1_kernel<<<BATCH, 256, 0, stream>>>(
        stu_id, exer_id, kp,
        student_mean, student_std_eta, k_diff_mean, k_diff_std_eta,
        e_disc_mean, e_disc_std_eta,
        lam1_stu, lam2_stu, lam1_exer, lam2_exer, stu_cnt, exer_cnt,
        w_mu1, w_eta1, b_mu1, b_eta1, x2);

    layer2_kernel<<<BATCH, 256, 0, stream>>>(x2, w_mu2, w_eta2, b_mu2, b_eta2, x3);

    layer3_kernel<<<BATCH, 256, 0, stream>>>(x3, w_mu3, w_eta3, b_mu3, b_eta3, out);

    // kl_w_sum -> slot 8192
    kl_gauss_kernel<<<1024, 256, 0, stream>>>(w_mu1, w_eta1, L1DIM * KDIM,  out + 8192);
    kl_gauss_kernel<<<256,  256, 0, stream>>>(w_mu2, w_eta2, L2DIM * L1DIM, out + 8192);
    kl_gauss_kernel<<<1,    256, 0, stream>>>(w_mu3, w_eta3, L2DIM,         out + 8192);
    // kl_b_sum -> slot 8193
    kl_gauss_kernel<<<1, 256, 0, stream>>>(b_mu1, b_eta1, L1DIM, out + 8193);
    kl_gauss_kernel<<<1, 256, 0, stream>>>(b_mu2, b_eta2, L2DIM, out + 8193);
    kl_gauss_kernel<<<1, 256, 0, stream>>>(b_mu3, b_eta3, 1,     out + 8193);
    // kl_diag -> slot 8194
    kl_diag_kernel<<<2048, 256, 0, stream>>>(student_mean, student_std_eta, stu_cnt,
                                             lam1_stu, lam2_stu, STU_N * KDIM, 1, out + 8194);
    kl_diag_kernel<<<2048, 256, 0, stream>>>(k_diff_mean, k_diff_std_eta, exer_cnt,
                                             lam1_exer, lam2_exer, EXER_N * KDIM, KDIM, out + 8194);
    kl_diag_kernel<<<128,  256, 0, stream>>>(e_disc_mean, e_disc_std_eta, exer_cnt,
                                             lam1_exer, lam2_exer, EXER_N, 1, out + 8194);
}
